// SE3Transformer_1271310320345
// MI455X (gfx1250) — compile-verified, matrix operand order checked by reading, not run on hardware
//
#include <hip/hip_runtime.h>
#include <hip/hip_bf16.h>
#include <math.h>

// ---------------------------------------------------------------------------
// SE(3)-Transformer forward for MI455X (gfx1250, wave32).
// Dense channel mixing -> v_wmma_f32_16x16x32_f16 (templated shapes, f16
// operands, fragment-order pre-swizzled B panels, b128 A loads).
// Edge softmax/aggregation -> two edge passes with L2-resident f32 atomics;
// softmax normalization applied as a separate cheap scaling pass.
// ---------------------------------------------------------------------------

typedef __attribute__((ext_vector_type(16))) _Float16 v16h;
typedef __attribute__((ext_vector_type(8)))  float    v8f;

#define SCALE_LOGIT 0.16222142113076254f   // 1/sqrt(38)

// monotonic float<->u32 encoding so atomicMax(u32) == float max
static __device__ __forceinline__ unsigned fenc(float x) {
  unsigned b = __float_as_uint(x);
  return (b & 0x80000000u) ? ~b : (b | 0x80000000u);
}
static __device__ __forceinline__ float fdec(unsigned u) {
  return __uint_as_float((u & 0x80000000u) ? (u ^ 0x80000000u) : ~u);
}
static __device__ __forceinline__ float dot4(float4 a, float4 b) {
  return a.x * b.x + a.y * b.y + a.z * b.z + a.w * b.w;
}

// ---------------- feature init: time embedding + degree features ------------
__global__ void se3_init_kernel(const float* __restrict__ nodef,
                                const float* __restrict__ t,
                                const float* __restrict__ vp,
                                float* __restrict__ te,
                                float* __restrict__ F0,
                                float* __restrict__ F1,
                                float* __restrict__ F2,
                                int NT, int Npg) {
  int n = blockIdx.x * blockDim.x + threadIdx.x;
  if (n >= NT) return;
  float tv = t[n / Npg];
#pragma unroll
  for (int i = 0; i < 16; ++i) {
    float f = __expf(-(float)i * 0.5756462732485114f);  // ln(1e4)/16
    float a = tv * f;
    te[(size_t)n * 32 + i]      = __sinf(a);
    te[(size_t)n * 32 + 16 + i] = __cosf(a);
  }
#pragma unroll
  for (int c = 0; c < 32; ++c) {
    F0[(size_t)n * 64 + c]      = 0.0f;                     // scalar channels
    F0[(size_t)n * 64 + 32 + c] = te[(size_t)n * 32 + c];   // time channels
  }
#pragma unroll
  for (int mm = 0; mm < 3; ++mm) {
    float x = nodef[(size_t)n * 3 + mm];
#pragma unroll
    for (int c = 0; c < 16; ++c) F1[(size_t)n * 48 + mm * 16 + c] = x * vp[c];
  }
#pragma unroll
  for (int i = 0; i < 40; ++i) F2[(size_t)n * 40 + i] = 0.0f;
}

// ---------------- pack f32 weights into fragment-ordered f16 B panels -------
// Fragment order: panel[((kc*NTILES + tn)*32 + lane)*16 + j] where
//   col = tn*16 + (lane&15)
//   kk  = kc*32 + ((j>>3)<<4) + ((lane>>4)<<3) + (j&7)
// so each wave lane loads its entire 16-half B fragment contiguously (32 B).
struct PackArgs {
  const float* Wq[15]; const float* Wk[15]; const float* Wv[15]; const float* Wo[15];
  _Float16* dQ[15];    _Float16* dO[15];
};

__global__ void se3_pack_weights(PackArgs pa) {
  const int C_[3]   = {64, 16, 8};
  const int HDK_[3] = {64, 16, 8};   // H*DK
  const int HDV_[3] = {32, 16, 8};   // H*dv
  const int NTQ_[3] = {10, 3, 2};    // QKV panel col-tiles
  const int KCHQ_[3] = {2, 1, 1};    // QKV panel K-chunks of 32
  const int NTO_[3] = {2, 1, 1};     // Wo panel col-tiles
  const int COUT_[3] = {32, 16, 8};
  int tid = blockIdx.x * blockDim.x + threadIdx.x;
  int stride = gridDim.x * blockDim.x;
  for (int li = 0; li < 5; ++li)
    for (int l = 0; l < 3; ++l) {
      int s = li * 3 + l;
      int C = C_[l], hdk = HDK_[l], hdv = HDV_[l];
      {
        int ntiles = NTQ_[l], kch = KCHQ_[l];
        int elems = kch * ntiles * 512;
        for (int i = tid; i < elems; i += stride) {
          int j = i & 15, lane = (i >> 4) & 31, r = i >> 9;
          int tn = r % ntiles, kc = r / ntiles;
          int colg = tn * 16 + (lane & 15);
          int kk = kc * 32 + ((j >> 3) << 4) + ((lane >> 4) << 3) + (j & 7);
          float v = 0.0f;
          if (kk < C) {
            if (colg < hdk)                 v = pa.Wq[s][kk * hdk + colg];
            else if (colg < 2 * hdk)        v = pa.Wk[s][kk * hdk + colg - hdk];
            else if (colg < 2 * hdk + hdv)  v = pa.Wv[s][kk * hdv + colg - 2 * hdk];
          }
          pa.dQ[s][i] = (_Float16)v;
        }
      }
      {
        int ntiles = NTO_[l], cout = COUT_[l];
        int elems = ntiles * 512;
        for (int i = tid; i < elems; i += stride) {
          int j = i & 15, lane = (i >> 4) & 31, tn = i >> 9;
          int colg = tn * 16 + (lane & 15);
          int kk = ((j >> 3) << 4) + ((lane >> 4) << 3) + (j & 7);
          float v = (kk < hdv && colg < cout) ? pa.Wo[s][kk * cout + colg] : 0.0f;
          pa.dO[s][i] = (_Float16)v;
        }
      }
    }
}

// ---------------- WMMA GEMM: C[RxW] = f16(A[RxLDA]) x Bpanel[KpadxW] --------
// K, LDA, NTILES compile-time: all bounds checks fold, K-loop fully unrolls,
// A loaded as 2x b128 per 8-element group, B fragment as one 32-byte vector.
template <int K, int LDA, int NTILES>
__global__ void se3_wmma_gemm(const float* __restrict__ A,
                              const _Float16* __restrict__ Bfrag,
                              float* __restrict__ C) {
  const int lane = threadIdx.x & 31;
  const bool hiHalf = lane >= 16;
  const int khi = hiHalf ? 8 : 0;
  const int row = blockIdx.x * 16 + (lane & 15);
  const int col = blockIdx.y * 16 + (lane & 15);
  constexpr int KCH = (K + 31) / 32;
  constexpr int W = NTILES * 16;   // panel width == output ldc
  v8f acc = {};
#pragma unroll
  for (int kc = 0; kc < KCH; ++kc) {
    v16h a;
#pragma unroll
    for (int g = 0; g < 2; ++g) {
      const int gb = kc * 32 + g * 16;            // constant after unroll
      const bool vLo = (gb + 8 <= K);             // compile-time
      const bool vHi = (gb + 16 <= K);            // compile-time
      const bool valid = hiHalf ? vHi : vLo;      // select of constants
      float4 lo = make_float4(0.f, 0.f, 0.f, 0.f);
      float4 hi = make_float4(0.f, 0.f, 0.f, 0.f);
      if (valid) {
        const float4* p = (const float4*)(A + (size_t)row * LDA + gb + khi);
        lo = p[0];
        hi = p[1];
      }
      a[g * 8 + 0] = (_Float16)lo.x; a[g * 8 + 1] = (_Float16)lo.y;
      a[g * 8 + 2] = (_Float16)lo.z; a[g * 8 + 3] = (_Float16)lo.w;
      a[g * 8 + 4] = (_Float16)hi.x; a[g * 8 + 5] = (_Float16)hi.y;
      a[g * 8 + 6] = (_Float16)hi.z; a[g * 8 + 7] = (_Float16)hi.w;
    }
    const v16h b = *(const v16h*)(
        Bfrag + (((size_t)kc * NTILES + blockIdx.y) * 32 + lane) * 16);
    acc = __builtin_amdgcn_wmma_f32_16x16x32_f16(false, a, false, b, (short)0,
                                                 acc, false, false);
  }
  const int rbase = blockIdx.x * 16 + (hiHalf ? 8 : 0);
#pragma unroll
  for (int i = 0; i < 8; ++i)
    C[(size_t)(rbase + i) * W + col] = acc[i];
}

// ---------------- edge pass 1: logits + running segment max -----------------
__global__ void se3_edge_logits(const int* __restrict__ edges, int E,
                                const float* __restrict__ eattr,
                                const float* __restrict__ We1,
                                const float* __restrict__ be1,
                                const float* __restrict__ We2,
                                const float* __restrict__ QKV0,
                                const float* __restrict__ QKV1,
                                const float* __restrict__ QKV2,
                                float* __restrict__ logits,
                                unsigned* __restrict__ nmax) {
  int e = blockIdx.x * blockDim.x + threadIdx.x;
  if (e >= E) return;
  int s = edges[e], d = edges[E + e];
  float acc[4] = {0.f, 0.f, 0.f, 0.f};
  {  // degree 0 (m=1, DK=16): 4 float4 per head
    const float4* q = (const float4*)(QKV0 + (size_t)d * 160);
    const float4* k = (const float4*)(QKV0 + (size_t)s * 160 + 64);
    __builtin_prefetch(QKV0 + (size_t)s * 160 + 128, 0, 3);  // V row for pass 2
#pragma unroll
    for (int h = 0; h < 4; ++h)
#pragma unroll
      for (int i = 0; i < 4; ++i) acc[h] += dot4(q[h * 4 + i], k[h * 4 + i]);
  }
#pragma unroll
  for (int mm = 0; mm < 3; ++mm) {  // degree 1 (DK=4): 1 float4 per head
    const float4* q = (const float4*)(QKV1 + ((size_t)d * 3 + mm) * 48);
    const float4* k = (const float4*)(QKV1 + ((size_t)s * 3 + mm) * 48 + 16);
#pragma unroll
    for (int h = 0; h < 4; ++h) acc[h] += dot4(q[h], k[h]);
  }
#pragma unroll
  for (int mm = 0; mm < 5; ++mm) {  // degree 2 (DK=2): 1 float2 per head
    const float2* q = (const float2*)(QKV2 + ((size_t)d * 5 + mm) * 32);
    const float2* k = (const float2*)(QKV2 + ((size_t)s * 5 + mm) * 32 + 8);
#pragma unroll
    for (int h = 0; h < 4; ++h)
      acc[h] += q[h].x * k[h].x + q[h].y * k[h].y;
  }
  float ea = eattr[e];
  float eh[16];
#pragma unroll
  for (int j = 0; j < 16; ++j) {
    float v = ea * We1[j] + be1[j];
    eh[j] = v > 0.f ? v : 0.f;
  }
#pragma unroll
  for (int h = 0; h < 4; ++h) {
    float el = 0.f;
#pragma unroll
    for (int j = 0; j < 16; ++j) el += eh[j] * We2[j * 4 + h];
    float lv = acc[h] * SCALE_LOGIT + el;
    logits[(size_t)e * 4 + h] = lv;
    atomicMax(&nmax[d * 4 + h], fenc(lv));
  }
}

// ---------------- edge pass 2: ex = exp(l - max); den += ex; agg += ex*V ----
__global__ void se3_edge_agg(const int* __restrict__ edges, int E,
                             const float* __restrict__ QKV0,
                             const float* __restrict__ QKV1,
                             const float* __restrict__ QKV2,
                             const float* __restrict__ logits,
                             const unsigned* __restrict__ nmax,
                             float* __restrict__ den,
                             float* __restrict__ agg0,
                             float* __restrict__ agg1,
                             float* __restrict__ agg2) {
  int e = blockIdx.x * blockDim.x + threadIdx.x;
  if (e >= E) return;
  int s = edges[e], d = edges[E + e];
  float ex[4];
#pragma unroll
  for (int h = 0; h < 4; ++h) {
    float lv = logits[(size_t)e * 4 + h];
    ex[h] = __expf(lv - fdec(nmax[d * 4 + h]));
    atomicAdd(&den[d * 4 + h], ex[h]);
  }
  {  // degree 0, dv=8: 2 float4 per head
    const float4* v = (const float4*)(QKV0 + (size_t)s * 160 + 128);
    float* a = agg0 + (size_t)d * 32;
#pragma unroll
    for (int h = 0; h < 4; ++h) {
      float4 va = v[h * 2], vb = v[h * 2 + 1];
      atomicAdd(&a[h * 8 + 0], ex[h] * va.x);
      atomicAdd(&a[h * 8 + 1], ex[h] * va.y);
      atomicAdd(&a[h * 8 + 2], ex[h] * va.z);
      atomicAdd(&a[h * 8 + 3], ex[h] * va.w);
      atomicAdd(&a[h * 8 + 4], ex[h] * vb.x);
      atomicAdd(&a[h * 8 + 5], ex[h] * vb.y);
      atomicAdd(&a[h * 8 + 6], ex[h] * vb.z);
      atomicAdd(&a[h * 8 + 7], ex[h] * vb.w);
    }
  }
#pragma unroll
  for (int mm = 0; mm < 3; ++mm) {  // degree 1, dv=4: 1 float4 per head
    const float4* v = (const float4*)(QKV1 + ((size_t)s * 3 + mm) * 48 + 32);
    float* a = agg1 + ((size_t)d * 3 + mm) * 16;
#pragma unroll
    for (int h = 0; h < 4; ++h) {
      float4 va = v[h];
      atomicAdd(&a[h * 4 + 0], ex[h] * va.x);
      atomicAdd(&a[h * 4 + 1], ex[h] * va.y);
      atomicAdd(&a[h * 4 + 2], ex[h] * va.z);
      atomicAdd(&a[h * 4 + 3], ex[h] * va.w);
    }
  }
#pragma unroll
  for (int mm = 0; mm < 5; ++mm) {  // degree 2, dv=2: 1 float2 per head
    const float2* v = (const float2*)(QKV2 + ((size_t)s * 5 + mm) * 32 + 16);
    float* a = agg2 + ((size_t)d * 5 + mm) * 8;
#pragma unroll
    for (int h = 0; h < 4; ++h) {
      float2 va = v[h];
      atomicAdd(&a[h * 2 + 0], ex[h] * va.x);
      atomicAdd(&a[h * 2 + 1], ex[h] * va.y);
    }
  }
}

// ---------------- softmax normalization: agg *= 1/(den+eps) -----------------
__global__ void se3_norm_agg(float* __restrict__ agg0, float* __restrict__ agg1,
                             float* __restrict__ agg2,
                             const float* __restrict__ den, int NT) {
  int n = blockIdx.x * blockDim.x + threadIdx.x;
  if (n >= NT) return;
  float inv[4];
#pragma unroll
  for (int h = 0; h < 4; ++h) inv[h] = 1.0f / (den[n * 4 + h] + 1e-9f);
#pragma unroll
  for (int c = 0; c < 32; ++c) agg0[(size_t)n * 32 + c] *= inv[c >> 3];
#pragma unroll
  for (int mm = 0; mm < 3; ++mm)
#pragma unroll
    for (int c = 0; c < 16; ++c)
      agg1[((size_t)n * 3 + mm) * 16 + c] *= inv[c >> 2];
#pragma unroll
  for (int mm = 0; mm < 5; ++mm)
#pragma unroll
    for (int c = 0; c < 8; ++c)
      agg2[((size_t)n * 5 + mm) * 8 + c] *= inv[c >> 1];
}

// ---------------- per-node: gating + te re-injection + equivariant RMSnorm --
__global__ void se3_post_kernel(const float* __restrict__ O0,
                                const float* __restrict__ O1,
                                const float* __restrict__ O2,
                                const float* __restrict__ te,
                                const float* __restrict__ Wg1,
                                const float* __restrict__ Wg2,
                                const float* __restrict__ g0,
                                const float* __restrict__ g1,
                                const float* __restrict__ g2,
                                float* __restrict__ F0,
                                float* __restrict__ F1,
                                float* __restrict__ F2, int NT) {
  int n = blockIdx.x * blockDim.x + threadIdx.x;
  if (n >= NT) return;
  float s0[32];
#pragma unroll
  for (int c = 0; c < 32; ++c) s0[c] = O0[(size_t)n * 32 + c];
  float gate1[16];
#pragma unroll
  for (int c = 0; c < 16; ++c) {
    float a = 0.f;
    for (int j = 0; j < 32; ++j) a += s0[j] * Wg1[j * 16 + c];
    gate1[c] = 1.f / (1.f + __expf(-a));
  }
  float gate2[8];
#pragma unroll
  for (int c = 0; c < 8; ++c) {
    float a = 0.f;
    for (int j = 0; j < 32; ++j) a += s0[j] * Wg2[j * 8 + c];
    gate2[c] = 1.f / (1.f + __expf(-a));
  }
  // degree 0: concat(out0, te) then RMS over 64 channels (m=1)
  float ssq = 0.f;
  for (int c = 0; c < 32; ++c) ssq += s0[c] * s0[c];
  for (int c = 0; c < 32; ++c) {
    float v = te[(size_t)n * 32 + c];
    ssq += v * v;
  }
  float r0 = rsqrtf(ssq * (1.f / 64.f) + 1e-6f);
  for (int c = 0; c < 32; ++c) F0[(size_t)n * 64 + c] = s0[c] * r0 * g0[c];
  for (int c = 0; c < 32; ++c)
    F0[(size_t)n * 64 + 32 + c] = te[(size_t)n * 32 + c] * r0 * g0[32 + c];
  // degree 1: gate then RMS over 16 channels of per-channel vector norms
  float ss1 = 0.f;
  for (int mm = 0; mm < 3; ++mm)
    for (int c = 0; c < 16; ++c) {
      float v = O1[((size_t)n * 3 + mm) * 16 + c] * gate1[c];
      ss1 += v * v;
    }
  float r1 = rsqrtf(ss1 * (1.f / 16.f) + 1e-6f);
  for (int mm = 0; mm < 3; ++mm)
    for (int c = 0; c < 16; ++c)
      F1[(size_t)n * 48 + mm * 16 + c] =
          O1[((size_t)n * 3 + mm) * 16 + c] * gate1[c] * r1 * g1[c];
  // degree 2
  float ss2 = 0.f;
  for (int mm = 0; mm < 5; ++mm)
    for (int c = 0; c < 8; ++c) {
      float v = O2[((size_t)n * 5 + mm) * 16 + c] * gate2[c];
      ss2 += v * v;
    }
  float r2 = rsqrtf(ss2 * (1.f / 8.f) + 1e-6f);
  for (int mm = 0; mm < 5; ++mm)
    for (int c = 0; c < 8; ++c)
      F2[(size_t)n * 40 + mm * 8 + c] =
          O2[((size_t)n * 5 + mm) * 16 + c] * gate2[c] * r2 * g2[c];
}

// ---------------- final projection: pos = feats1 @ op ----------------------
__global__ void se3_final_kernel(const float* __restrict__ O1,
                                 const float* __restrict__ opW,
                                 float* __restrict__ out, int NT) {
  int n = blockIdx.x * blockDim.x + threadIdx.x;
  if (n >= NT) return;
#pragma unroll
  for (int mm = 0; mm < 3; ++mm) {
    float a = 0.f;
#pragma unroll
    for (int c = 0; c < 16; ++c) a += O1[((size_t)n * 3 + mm) * 16 + c] * opW[c];
    out[(size_t)n * 3 + mm] = a;
  }
}

// ---------------------------------------------------------------------------
extern "C" void kernel_launch(void* const* d_in, const int* in_sizes, int n_in,
                              void* d_out, int out_size, void* d_ws, size_t ws_size,
                              hipStream_t stream) {
  (void)n_in; (void)out_size; (void)ws_size;
  const float* nodef = (const float*)d_in[0];   // [B, N, 3]
  const float* tptr  = (const float*)d_in[1];   // [B]
  const int*   edges = (const int*)d_in[2];     // [2, E]
  const float* eattr = (const float*)d_in[3];   // [E, 1]
  const int NT  = in_sizes[0] / 3;
  const int E   = in_sizes[3];
  const int Npg = 2048;                         // nodes per graph (L*BB)

  // --- params: jax pytree (sorted dict key) flatten order ---
  struct HostLayer {
    const float *We1, *We2, *Wg1, *Wg2, *be1;
    const float *Wk[3], *Wo[3], *Wq[3], *Wv[3];
  };
  int idx = 4;
  HostLayer L[5];
  auto readLayer = [&](HostLayer& ly) {
    ly.We1 = (const float*)d_in[idx++];
    ly.We2 = (const float*)d_in[idx++];
    ly.Wg1 = (const float*)d_in[idx++];
    ly.Wg2 = (const float*)d_in[idx++];
    ly.be1 = (const float*)d_in[idx++];
    for (int l = 0; l < 3; ++l) {   // keys sorted: Wk < Wo < Wq < Wv
      ly.Wk[l] = (const float*)d_in[idx++];
      ly.Wo[l] = (const float*)d_in[idx++];
      ly.Wq[l] = (const float*)d_in[idx++];
      ly.Wv[l] = (const float*)d_in[idx++];
    }
  };
  for (int i = 0; i < 4; ++i) readLayer(L[i]);       // 'layers'
  const float* normg[3];
  for (int l = 0; l < 3; ++l) normg[l] = (const float*)d_in[idx++];  // 'norm'
  const float* opW = (const float*)d_in[idx++];      // 'op'
  readLayer(L[4]);                                   // 'out'
  const float* vp = (const float*)d_in[idx++];       // 'vp'

  // --- workspace carve-up ---
  char* w = (char*)d_ws;
  size_t off = 0;
  auto carve = [&](size_t bytes) -> void* {
    void* p = w + off;
    off = (off + bytes + 255) & ~(size_t)255;
    return p;
  };
  float* te   = (float*)carve((size_t)NT * 32 * 4);
  float* F0   = (float*)carve((size_t)NT * 64 * 4);
  float* F1   = (float*)carve((size_t)NT * 48 * 4);
  float* F2   = (float*)carve((size_t)NT * 40 * 4);
  float* QKV0 = (float*)carve((size_t)NT * 160 * 4);
  float* QKV1 = (float*)carve((size_t)NT * 3 * 48 * 4);
  float* QKV2 = (float*)carve((size_t)NT * 5 * 32 * 4);
  float* lgts = (float*)carve((size_t)E * 4 * 4);
  // contiguous zero-region: nmax | den | agg0 | agg1 | agg2
  size_t nmaxB = (size_t)NT * 4 * 4, denB = (size_t)NT * 4 * 4;
  size_t a0B = (size_t)NT * 32 * 4, a1B = (size_t)NT * 48 * 4, a2B = (size_t)NT * 40 * 4;
  size_t ztot = nmaxB + denB + a0B + a1B + a2B;
  char* zbase = (char*)carve(ztot);
  unsigned* nmax = (unsigned*)zbase;
  float* den  = (float*)(zbase + nmaxB);
  float* agg0 = (float*)(zbase + nmaxB + denB);
  float* agg1 = (float*)(zbase + nmaxB + denB + a0B);
  float* agg2 = (float*)(zbase + nmaxB + denB + a0B + a1B);
  float* O0 = (float*)carve((size_t)NT * 32 * 4);
  float* O1 = (float*)carve((size_t)NT * 3 * 16 * 4);
  float* O2 = (float*)carve((size_t)NT * 5 * 16 * 4);
  // packed fragment-order f16 weight panels
  const int qkvElems[3] = {2 * 10 * 512, 1 * 3 * 512, 1 * 2 * 512};
  const int oElems[3]   = {2 * 512, 1 * 512, 1 * 512};
  _Float16* pQ[15];
  _Float16* pO[15];
  for (int li = 0; li < 5; ++li)
    for (int l = 0; l < 3; ++l) {
      pQ[li * 3 + l] = (_Float16*)carve((size_t)qkvElems[l] * 2);
      pO[li * 3 + l] = (_Float16*)carve((size_t)oElems[l] * 2);
    }

  // --- launch: init + weight pack ---
  se3_init_kernel<<<(NT + 255) / 256, 256, 0, stream>>>(nodef, tptr, vp, te, F0,
                                                        F1, F2, NT, Npg);
  PackArgs pa;
  for (int li = 0; li < 5; ++li)
    for (int l = 0; l < 3; ++l) {
      int s = li * 3 + l;
      pa.Wq[s] = L[li].Wq[l]; pa.Wk[s] = L[li].Wk[l];
      pa.Wv[s] = L[li].Wv[l]; pa.Wo[s] = L[li].Wo[l];
      pa.dQ[s] = pQ[s];       pa.dO[s] = pO[s];
    }
  se3_pack_weights<<<32, 256, 0, stream>>>(pa);

  const int eg = (E + 255) / 256;
  const int ng = (NT + 255) / 256;
  for (int li = 0; li < 5; ++li) {
    // QKV projections (WMMA): one panel per degree produces Q|K|V
    se3_wmma_gemm<64, 64, 10><<<dim3(NT / 16, 10), 32, 0, stream>>>(
        F0, pQ[li * 3 + 0], QKV0);
    se3_wmma_gemm<16, 16, 3><<<dim3(NT * 3 / 16, 3), 32, 0, stream>>>(
        F1, pQ[li * 3 + 1], QKV1);
    se3_wmma_gemm<8, 8, 2><<<dim3(NT * 5 / 16, 2), 32, 0, stream>>>(
        F2, pQ[li * 3 + 2], QKV2);
    // zero nmax (0 == encoded identity for max), den, aggs in one memset
    hipMemsetAsync(zbase, 0, ztot, stream);
    // edge passes
    se3_edge_logits<<<eg, 256, 0, stream>>>(edges, E, eattr, L[li].We1,
                                            L[li].be1, L[li].We2, QKV0, QKV1,
                                            QKV2, lgts, nmax);
    se3_edge_agg<<<eg, 256, 0, stream>>>(edges, E, QKV0, QKV1, QKV2, lgts,
                                         nmax, den, agg0, agg1, agg2);
    // softmax normalization, then output projections (WMMA)
    se3_norm_agg<<<ng, 256, 0, stream>>>(agg0, agg1, agg2, den, NT);
    se3_wmma_gemm<32, 32, 2><<<dim3(NT / 16, 2), 32, 0, stream>>>(
        agg0, pO[li * 3 + 0], O0);
    se3_wmma_gemm<16, 16, 1><<<dim3(NT * 3 / 16, 1), 32, 0, stream>>>(
        agg1, pO[li * 3 + 1], O1);
    se3_wmma_gemm<8, 8, 1><<<dim3(NT * 5 / 16, 1), 32, 0, stream>>>(
        agg2, pO[li * 3 + 2], O2);
    if (li < 4) {
      se3_post_kernel<<<ng, 256, 0, stream>>>(
          O0, O1, O2, te, L[li].Wg1, L[li].Wg2, normg[0], normg[1], normg[2],
          F0, F1, F2, NT);
    } else {
      se3_final_kernel<<<ng, 256, 0, stream>>>(O1, opW, (float*)d_out, NT);
    }
  }
}
